// GraphGRU_14946486190826
// MI455X (gfx1250) — matrix operand order repaired and checked by reading, not compiled, not run on hardware
//
#include <hip/hip_runtime.h>
#include <hip/hip_bf16.h>

// ---------------------------------------------------------------------------
// GraphGRU on MI455X (gfx1250), wave32 + WMMA bf16.
//
//   precompute:  XZ = x@Wz_top + bz,  XH = x@Wh_top + bh,  R1 = x@Wr + br
//   per step:
//     hUr  = h @ Ur                    (gather commutes with matmul: /8 FLOPs)
//     sum_h[n] = sum_k h[g[n,k]]
//     sumg[n]  = sum_k sigmoid(R1[n] + hUr[g[n,k]]) * h[g[n,k]]
//     z    = sigmoid(XZ + sum_h @ Wz_bot)
//     h    = ((1-z)*sum_h + z*tanh(XH + sumg @ Wh_bot)) * mask
// ---------------------------------------------------------------------------

typedef __attribute__((ext_vector_type(16))) __bf16 v16bf;
typedef __attribute__((ext_vector_type(8)))  float  v8f;

union BfFrag { v16bf v; uint4 q[2]; };

__device__ __forceinline__ unsigned short f2bf(float f) {
  unsigned int u = __float_as_uint(f);
  unsigned int r = u + 0x7FFFu + ((u >> 16) & 1u);   // round-to-nearest-even
  return (unsigned short)(r >> 16);
}

__device__ __forceinline__ float sigmoidf_(float x) {
  return 1.0f / (1.0f + __expf(-x));
}

// ---------------------------------------------------------------------------
// GEMM: C[M,256] = A[M,256](bf16 row-major) @ B[256,256], B given transposed
// (BT[n][k], bf16). Block = 256 threads = 8 waves; computes 256 rows x 128
// cols (grid.y selects column half). BT half staged to 64KB LDS.
// Each wave owns TWO 16-row tiles so every B fragment feeds 2 WMMAs, and the
// B fragment stream is software-pipelined to hide ds latency.
// MODE 0: out32 = C + bias[col]                (bias may be null)
// MODE 1: out32 = sigmoid(C + add)             (z gate)
// MODE 2: pre   = tanh(C + add);  h = ((1-z)*sum_h + z*pre)*mask
//         out32 = h;  outb = bf16(h)
// ---------------------------------------------------------------------------
template <int MODE>
__global__ __launch_bounds__(256) void gemm256(
    const unsigned short* __restrict__ A,
    const unsigned short* __restrict__ BT,
    const float* __restrict__ bias,
    const float* __restrict__ add,
    const float* __restrict__ zbuf,
    const float* __restrict__ sumh,
    float* __restrict__ out32,
    unsigned short* __restrict__ outb,
    int Mreal)
{
  __shared__ unsigned short ldsBT[128 * 256];   // 64 KB

  const int tid   = threadIdx.x;
  const int cbase = blockIdx.y * 128;

  // Stage transposed-B column-half into LDS (contiguous, coalesced).
  {
    const uint4* src = reinterpret_cast<const uint4*>(BT + (size_t)cbase * 256);
    uint4*       dst = reinterpret_cast<uint4*>(ldsBT);
#pragma unroll
    for (int i = 0; i < 16; ++i) dst[tid + 256 * i] = src[tid + 256 * i];
  }
  __syncthreads();

  const int wave = tid >> 5;
  const int lane = tid & 31;
  const int half = lane >> 4;    // 0: low 16 lanes, 1: high 16 lanes
  const int l16  = lane & 15;
  const int rowBase = blockIdx.x * 256 + wave * 32;   // 2 row tiles per wave

  v8f acc[2][8];
#pragma unroll
  for (int t = 0; t < 2; ++t)
#pragma unroll
    for (int j = 0; j < 8; ++j) { v8f zed = {}; acc[t][j] = zed; }

  // A fragment bases per ISA 16-bit A layout:
  // lane<16: K {0..7}+{16..23}; lane>=16: K {8..15}+{24..31}
  const unsigned short* aptr0 = A + (size_t)(rowBase + l16) * 256 + half * 8;
  const unsigned short* aptr1 = aptr0 + 16 * 256;

  const unsigned short* bbase = ldsBT + (size_t)l16 * 256 + half * 8;

#pragma unroll
  for (int kc = 0; kc < 8; ++kc) {
    BfFrag fa0, fa1;
    fa0.q[0] = *reinterpret_cast<const uint4*>(aptr0 + kc * 32);
    fa0.q[1] = *reinterpret_cast<const uint4*>(aptr0 + kc * 32 + 16);
    fa1.q[0] = *reinterpret_cast<const uint4*>(aptr1 + kc * 32);
    fa1.q[1] = *reinterpret_cast<const uint4*>(aptr1 + kc * 32 + 16);

    // Software-pipelined B fragment stream over the 8 column tiles.
    BfFrag fbc, fbn;
    {
      const unsigned short* bp = bbase + kc * 32;           // j = 0
      fbc.q[0] = *reinterpret_cast<const uint4*>(bp);
      fbc.q[1] = *reinterpret_cast<const uint4*>(bp + 16);
    }
#pragma unroll
    for (int j = 0; j < 8; ++j) {
      if (j < 7) {
        const unsigned short* bp = bbase + (size_t)(j + 1) * 16 * 256 + kc * 32;
        fbn.q[0] = *reinterpret_cast<const uint4*>(bp);
        fbn.q[1] = *reinterpret_cast<const uint4*>(bp + 16);
      }
      acc[0][j] = __builtin_amdgcn_wmma_f32_16x16x32_bf16(
          false, fa0.v, false, fbc.v, (short)0, acc[0][j], false, false);
      acc[1][j] = __builtin_amdgcn_wmma_f32_16x16x32_bf16(
          false, fa1.v, false, fbc.v, (short)0, acc[1][j], false, false);
      fbc = fbn;
    }
  }

  // Epilogue. C layout: VGPR g, lane l -> row = g + 8*half, col = l16.
#pragma unroll
  for (int t = 0; t < 2; ++t) {
#pragma unroll
    for (int j = 0; j < 8; ++j) {
      const int col = cbase + j * 16 + l16;
#pragma unroll
      for (int g = 0; g < 8; ++g) {
        const int row = rowBase + t * 16 + half * 8 + g;
        if (row < Mreal) {
          const size_t idx = (size_t)row * 256 + col;
          float v = acc[t][j][g];
          if (MODE == 0) {
            if (bias != nullptr) v += bias[col];
            out32[idx] = v;
          } else if (MODE == 1) {
            out32[idx] = sigmoidf_(v + add[idx]);
          } else {
            const float pre = tanhf(v + add[idx]);
            const float zv  = zbuf[idx];
            const float sh  = sumh[idx];
            float hv = (1.0f - zv) * sh + zv * pre;
            if (row == 0) hv = 0.0f;             // null-message row mask
            out32[idx] = hv;
            outb[idx]  = f2bf(hv);
          }
        }
      }
    }
  }
}

// ---------------------------------------------------------------------------
// Gather + gate: one block per node, one thread per hidden channel.
// h and hUr are L2-resident (102 MB each vs 192 MB L2).
// ---------------------------------------------------------------------------
__global__ __launch_bounds__(256) void gather_sum(
    const int* __restrict__ gidx,
    const float* __restrict__ h,
    const float* __restrict__ hUr,
    const float* __restrict__ R1,
    float* __restrict__ sum_h,
    unsigned short* __restrict__ sumh_b,
    unsigned short* __restrict__ sumg_b)
{
  const int n = blockIdx.x;
  const int c = threadIdx.x;
  const size_t nc = (size_t)n * 256 + c;
  const float r1 = R1[nc];
  float s = 0.0f, sg = 0.0f;
#pragma unroll
  for (int k = 0; k < 8; ++k) {
    const int j = gidx[n * 8 + k];
    const size_t o = (size_t)j * 256 + c;
    const float hv = h[o];
    const float r  = sigmoidf_(r1 + hUr[o]);
    s  += hv;
    sg += r * hv;
  }
  sum_h[nc]  = s;
  sumh_b[nc] = f2bf(s);
  sumg_b[nc] = f2bf(sg);
}

// Convert + transpose + split weights to bf16 (done once, tiny).
// Output order: [WzTopT, WzBotT, WrT, UrT, WhTopT, WhBotT], each T[n*256+k].
__global__ __launch_bounds__(256) void wprep(
    const float* __restrict__ Wz, const float* __restrict__ Wr,
    const float* __restrict__ Ur, const float* __restrict__ Wh,
    unsigned short* __restrict__ out)
{
  const int idx = blockIdx.x * 256 + threadIdx.x;   // 6*65536 total
  const int m  = idx >> 16;
  const int r  = idx & 0xFFFF;
  const int nn = r >> 8;
  const int kk = r & 255;
  float v;
  switch (m) {
    case 0:  v = Wz[kk * 256 + nn];         break;
    case 1:  v = Wz[(256 + kk) * 256 + nn]; break;
    case 2:  v = Wr[kk * 256 + nn];         break;
    case 3:  v = Ur[kk * 256 + nn];         break;
    case 4:  v = Wh[kk * 256 + nn];         break;
    default: v = Wh[(256 + kk) * 256 + nn]; break;
  }
  out[idx] = f2bf(v);
}

__global__ __launch_bounds__(256) void conv_x(
    const float* __restrict__ x, unsigned short* __restrict__ xb, int total)
{
  const int i = blockIdx.x * 256 + threadIdx.x;
  if (i < total) xb[i] = f2bf(x[i]);
}

__global__ __launch_bounds__(256) void conv_h(
    const float* __restrict__ h, float* __restrict__ h_cur,
    unsigned short* __restrict__ hb, int total)
{
  const int i = blockIdx.x * 256 + threadIdx.x;
  if (i < total) { const float v = h[i]; h_cur[i] = v; hb[i] = f2bf(v); }
}

// ---------------------------------------------------------------------------
extern "C" void kernel_launch(void* const* d_in, const int* in_sizes, int n_in,
                              void* d_out, int out_size, void* d_ws, size_t ws_size,
                              hipStream_t stream)
{
  const float* h0 = (const float*)d_in[0];
  const float* x  = (const float*)d_in[1];
  const int*   mg = (const int*)d_in[2];
  const float* Wz = (const float*)d_in[3];
  const float* bz = (const float*)d_in[4];
  const float* Wr = (const float*)d_in[5];
  const float* Ur = (const float*)d_in[6];
  const float* br = (const float*)d_in[7];
  const float* Wh = (const float*)d_in[8];
  const float* bh = (const float*)d_in[9];
  float* out = (float*)d_out;

  const int N  = in_sizes[0] / 256;
  const int NP = ((N + 255) / 256) * 256;          // pad rows to block tile
  const size_t S32 = (size_t)NP * 256 * sizeof(float);
  const size_t S16 = (size_t)NP * 256 * sizeof(unsigned short);

  char* p = (char*)d_ws;
  float* h_cur = (float*)p;  p += S32;
  float* hUr   = (float*)p;  p += S32;
  float* R1    = (float*)p;  p += S32;
  float* XZ    = (float*)p;  p += S32;
  float* XH    = (float*)p;  p += S32;
  float* sum_h = (float*)p;  p += S32;
  float* zbuf  = (float*)p;  p += S32;
  unsigned short* hb     = (unsigned short*)p; p += S16;
  unsigned short* xb     = (unsigned short*)p; p += S16;
  unsigned short* sumh_b = (unsigned short*)p; p += S16;
  unsigned short* sumg_b = (unsigned short*)p; p += S16;
  unsigned short* wbuf   = (unsigned short*)p; p += 6 * 65536 * sizeof(unsigned short);

  const unsigned short* WzTopT = wbuf + 0 * 65536;
  const unsigned short* WzBotT = wbuf + 1 * 65536;
  const unsigned short* WrT    = wbuf + 2 * 65536;
  const unsigned short* UrT    = wbuf + 3 * 65536;
  const unsigned short* WhTopT = wbuf + 4 * 65536;
  const unsigned short* WhBotT = wbuf + 5 * 65536;

  const dim3 gGemm(NP / 256, 2);
  const dim3 b256(256);
  const int total = N * 256;
  const int cgrid = (total + 255) / 256;

  wprep<<<(6 * 65536) / 256, b256, 0, stream>>>(Wz, Wr, Ur, Wh, wbuf);
  conv_x<<<cgrid, b256, 0, stream>>>(x, xb, total);
  conv_h<<<cgrid, b256, 0, stream>>>(h0, h_cur, hb, total);

  // Loop-invariant halves: XZ = x@WzTop + bz, XH = x@WhTop + bh, R1 = x@Wr + br
  gemm256<0><<<gGemm, b256, 0, stream>>>(xb, WzTopT, bz, nullptr, nullptr, nullptr, XZ, nullptr, N);
  gemm256<0><<<gGemm, b256, 0, stream>>>(xb, WhTopT, bh, nullptr, nullptr, nullptr, XH, nullptr, N);
  gemm256<0><<<gGemm, b256, 0, stream>>>(xb, WrT,    br, nullptr, nullptr, nullptr, R1, nullptr, N);

  for (int d = 0; d < 6; ++d) {
    // hUr = h @ Ur
    gemm256<0><<<gGemm, b256, 0, stream>>>(hb, UrT, nullptr, nullptr, nullptr, nullptr, hUr, nullptr, N);
    // neighborhood reductions (L2-resident gathers)
    gather_sum<<<N, b256, 0, stream>>>(mg, h_cur, hUr, R1, sum_h, sumh_b, sumg_b);
    // z = sigmoid(XZ + sum_h @ WzBot)
    gemm256<1><<<gGemm, b256, 0, stream>>>(sumh_b, WzBotT, nullptr, XZ, nullptr, nullptr, zbuf, nullptr, N);
    // h = ((1-z)*sum_h + z*tanh(XH + sumg @ WhBot)) * mask ; emit f32 + bf16
    float* hout = (d == 5) ? out : h_cur;
    gemm256<2><<<gGemm, b256, 0, stream>>>(sumg_b, WhBotT, nullptr, XH, zbuf, sum_h, hout, hb, N);
  }
}
